// ChaoticGRU_27101243638132
// MI455X (gfx1250) — matrix operand order repaired and checked by reading, not compiled, not run on hardware
//
#include <hip/hip_runtime.h>
#include <math.h>

// ---------------------------------------------------------------------------
// Bidirectional GRU for MI455X (gfx1250), wave32 + WMMA bf16 16x16x32.
//
// Persistent scan kernel: 128 blocks x 64 threads (2 waves). Block owns one
// (direction, 16-column tile); its two waves cover the two 16-row batch
// tiles. The block's full weight slice (r/z/n columns, K=512 input-side +
// K=1024 recurrent-side, bf16) is staged into LDS once (~149 KB, fits the
// 320 KB WGP LDS at 2 blocks/WGP) and re-read every step via ds_load_b128.
// h_prev for the blend lives in registers (the D-tile of step t is the
// blend input of step t+1 in identical lane positions); only the bf16
// mirror of h (WMMA A operand, read cross-block) goes through memory,
// double-buffered, with one per-direction grid barrier per step.
// ---------------------------------------------------------------------------

typedef __attribute__((ext_vector_type(16))) __bf16 v16bf;
typedef __attribute__((ext_vector_type(8)))  __bf16 v8bf;
typedef __attribute__((ext_vector_type(8)))  float  v8f;

#define GB   32      // batch
#define GS   1024    // seq
#define GK   512     // input features
#define GH   1024    // hidden

// LDS layout: 48 rows (3 groups x 16 cols) of recurrent K=1024 and input
// K=512 weights, rows padded so 16 column-lanes hit distinct bank groups.
#define WRECP 1032   // 1024 + 8  (2064 B/row, 16B aligned, stride%64dw = 4)
#define WINPP 520    // 512  + 8  (1040 B/row)
#define LDS_ELEMS (48 * WRECP + 48 * WINPP)   // 74,496 bf16 = 148,992 B

// ---- workspace layout (bytes) ----
static const size_t SZ_XBF   = (size_t)GB*GS*GK*2;          // x in bf16
static const size_t SZ_WINP  = (size_t)3*GH*GK*2;           // [3072][512] bf16
static const size_t SZ_WREC  = (size_t)3*GH*GH*2;           // [3072][1024] bf16
static const size_t SZ_HBF   = (size_t)2*2*GB*GH*2;         // 2 bufs x 2 dirs bf16
static const size_t OFF_XBF  = 0;
static const size_t OFF_WIF  = OFF_XBF + SZ_XBF;
static const size_t OFF_WRF  = OFF_WIF + SZ_WINP;
static const size_t OFF_WIB  = OFF_WRF + SZ_WREC;
static const size_t OFF_WRB  = OFF_WIB + SZ_WINP;
static const size_t OFF_HBF  = OFF_WRB + SZ_WREC;
static const size_t OFF_BAR  = OFF_HBF + SZ_HBF;            // unsigned[64]

// ---------------------------------------------------------------------------
__global__ void gru_cvt_x(const float* __restrict__ x, __bf16* __restrict__ xbf,
                          size_t n) {
  size_t i = (size_t)blockIdx.x * blockDim.x + threadIdx.x;
  size_t stride = (size_t)gridDim.x * blockDim.x;
  for (; i < n; i += stride) xbf[i] = (__bf16)x[i];
}

// dst[n*K + k] = bf16(src[k*ld + n])   (transpose-pack to column-major bf16)
__global__ void gru_pack_w(const float* __restrict__ src, __bf16* __restrict__ dst,
                           int K, int N, int ld) {
  size_t total = (size_t)K * N;
  size_t i = (size_t)blockIdx.x * blockDim.x + threadIdx.x;
  size_t stride = (size_t)gridDim.x * blockDim.x;
  for (; i < total; i += stride) {
    int n = (int)(i / K);
    int k = (int)(i % K);
    dst[i] = (__bf16)src[(size_t)k * ld + n];
  }
}

__global__ void gru_init(__bf16* __restrict__ hbf, unsigned* __restrict__ bar) {
  size_t n = (size_t)2*2*GB*GH;
  size_t i = (size_t)blockIdx.x * blockDim.x + threadIdx.x;
  size_t stride = (size_t)gridDim.x * blockDim.x;
  for (size_t j = i; j < n; j += stride) hbf[j] = (__bf16)0.0f;
  if (i < 64) bar[i] = 0u;
}

// ---------------------------------------------------------------------------
// A fragment (16-bit A 16x32): lane holds M = lane%16; element e has
// K = e + 16*(lane>=16). p already includes the +16*(lane>=16).
__device__ __forceinline__ v16bf load_afrag(const __bf16* p) {
  v8bf lo = *(const v8bf*)(p);
  v8bf hi = *(const v8bf*)(p + 8);
  v16bf a;
#pragma unroll
  for (int i = 0; i < 8; ++i) { a[i] = lo[i]; a[i + 8] = hi[i]; }
  return a;
}

// B fragment (16-bit B 32x16): lane holds N = lane%16; element e has
// K = e + 16*(lane>=16); p includes the +16*(lane>=16). Two 16B LDS loads.
__device__ __forceinline__ v16bf load_bfrag(const __bf16* p) {
  v8bf lo = *(const v8bf*)(p);
  v8bf hi = *(const v8bf*)(p + 8);
  v16bf b;
#pragma unroll
  for (int i = 0; i < 8; ++i) { b[i] = lo[i]; b[i + 8] = hi[i]; }
  return b;
}

__device__ __forceinline__ v8f wmma_bf16(v16bf a, v16bf b, v8f c) {
  return __builtin_amdgcn_wmma_f32_16x16x32_bf16(false, a, false, b,
                                                 (short)0, c, false, false);
}

__device__ __forceinline__ float sigmoidf_(float x) {
  return 1.0f / (1.0f + __expf(-x));
}

// ---------------------------------------------------------------------------
__global__ void __launch_bounds__(64, 1)
gru_scan(const __bf16* __restrict__ xbf,
         const __bf16* __restrict__ winp_f, const __bf16* __restrict__ wrec_f,
         const __bf16* __restrict__ winp_b, const __bf16* __restrict__ wrec_b,
         const float* __restrict__ Bg_f,  const float* __restrict__ Bni_f,
         const float* __restrict__ Bnh_f, const float* __restrict__ Bg_b,
         const float* __restrict__ Bni_b, const float* __restrict__ Bnh_b,
         __bf16* __restrict__ hbf, unsigned* __restrict__ bar,
         float* __restrict__ out, float* __restrict__ ht) {
  extern __shared__ __bf16 smem[];
  __bf16* lrec = smem;               // 48 rows x WRECP
  __bf16* linp = smem + 48 * WRECP;  // 48 rows x WINPP

  const int tid   = threadIdx.x;     // 0..63
  const int lane  = tid & 31;
  const int mtile = tid >> 5;        // wave in block = batch row tile (0/1)
  const int dir   = blockIdx.x >> 6; // 0 fwd, 1 bwd
  const int j0    = (blockIdx.x & 63) << 4;   // column tile base in [0,1024)
  const int half  = lane >> 4;
  const int lcol  = lane & 15;
  const int j     = j0 + lcol;

  const __bf16* Winp = dir ? winp_b : winp_f;
  const __bf16* Wrec = dir ? wrec_b : wrec_f;
  const float*  Bgp  = dir ? Bg_b  : Bg_f;
  const float*  Bnip = dir ? Bni_b : Bni_f;
  const float*  Bnhp = dir ? Bnh_b : Bnh_f;

  // ---- stage this block's weight slice into LDS (once) ----
  for (int u = tid; u < 48 * (GH / 8); u += 64) {   // recurrent side
    const int rg = u >> 7;                          // row 0..47
    const int ck = (u & 127) << 3;                  // K chunk base
    const int g = rg >> 4, c = rg & 15;
    *(v8bf*)(lrec + rg * WRECP + ck) =
        *(const v8bf*)(Wrec + ((size_t)(g * GH + j0 + c)) * GH + ck);
  }
  for (int u = tid; u < 48 * (GK / 8); u += 64) {   // input side
    const int rg = u >> 6;
    const int ck = (u & 63) << 3;
    const int g = rg >> 4, c = rg & 15;
    *(v8bf*)(linp + rg * WINPP + ck) =
        *(const v8bf*)(Winp + ((size_t)(g * GH + j0 + c)) * GK + ck);
  }
  __syncthreads();

  // Per-lane bias for this lane's D column (N = lcol)
  const float b_r  = Bgp[j];
  const float b_z  = Bgp[GH + j];
  const float b_nx = Bnip[j];
  const float b_nh = Bnhp[j];

  // LDS B-fragment bases (row = group*16 + lcol, +16*half for lane half)
  const __bf16* li_r = linp + (size_t)( 0 + lcol) * WINPP + 16 * half;
  const __bf16* li_z = linp + (size_t)(16 + lcol) * WINPP + 16 * half;
  const __bf16* li_n = linp + (size_t)(32 + lcol) * WINPP + 16 * half;
  const __bf16* lr_r = lrec + (size_t)( 0 + lcol) * WRECP + 16 * half;
  const __bf16* lr_z = lrec + (size_t)(16 + lcol) * WRECP + 16 * half;
  const __bf16* lr_n = lrec + (size_t)(32 + lcol) * WRECP + 16 * half;

  const int arow = mtile * 16 + lcol;    // A-fragment batch row for this lane
  unsigned* cnt  = bar + dir * 32;       // per-direction barrier counter

  v8f hreg = {};                         // h_prev for this wave's D elements

  for (int t = 0; t < GS; ++t) {
    const int cur = t & 1, nxt = cur ^ 1;
    const int s = dir ? (GS - 1 - t) : t;

    const __bf16* hcurbf = hbf + ((size_t)cur * 2 + dir) * (GB * GH);
    __bf16*       hnxtbf = hbf + ((size_t)nxt * 2 + dir) * (GB * GH);

    v8f accR = {}, accZ = {}, accNX = {}, accNH = {};

    // ---- x_t contribution: K = 512 (A from global/L2, B from LDS) ----
    const __bf16* ax = xbf + ((size_t)arow * GS + s) * GK + 16 * half;
#pragma unroll 4
    for (int kk = 0; kk < GK; kk += 32) {
      v16bf a = load_afrag(ax + kk);
      accR  = wmma_bf16(a, load_bfrag(li_r + kk), accR);
      accZ  = wmma_bf16(a, load_bfrag(li_z + kk), accZ);
      accNX = wmma_bf16(a, load_bfrag(li_n + kk), accNX);
    }

    // ---- h contribution: K = 1024 ----
    const __bf16* ah = hcurbf + (size_t)arow * GH + 16 * half;
#pragma unroll 4
    for (int kk = 0; kk < GH; kk += 32) {
      v16bf a = load_afrag(ah + kk);
      accR  = wmma_bf16(a, load_bfrag(lr_r + kk), accR);
      accZ  = wmma_bf16(a, load_bfrag(lr_z + kk), accZ);
      accNH = wmma_bf16(a, load_bfrag(lr_n + kk), accNH);
    }

    // ---- elementwise GRU update (D layout: M = v + 8*half, N = lcol) ----
#pragma unroll
    for (int v = 0; v < 8; ++v) {
      const int b  = mtile * 16 + v + 8 * half;   // batch row of D element
      const float gr = sigmoidf_(accR[v] + b_r);
      const float gz = sigmoidf_(accZ[v] + b_z);
      const float nn = tanhf((accNX[v] + b_nx) + gr * (accNH[v] + b_nh));
      const float hn = (1.0f - gz) * nn + gz * hreg[v];
      hreg[v] = hn;
      hnxtbf[(size_t)b * GH + j] = (__bf16)hn;
      out[((size_t)b * GS + s) * (2 * GH) + (size_t)dir * GH + j] = hn;
      if (t == GS - 1)
        ht[(size_t)b * (2 * GH) + (size_t)dir * GH + j] = hn;
    }

    // ---- one per-direction grid barrier per step (64 blocks/dir) ----
    __threadfence();
    __syncthreads();
    if (tid == 0) {
      __hip_atomic_fetch_add(cnt, 1u, __ATOMIC_RELEASE,
                             __HIP_MEMORY_SCOPE_AGENT);
      const unsigned target = (unsigned)(t + 1) * 64u;
      while (__hip_atomic_load(cnt, __ATOMIC_ACQUIRE,
                               __HIP_MEMORY_SCOPE_AGENT) < target) {
        __builtin_amdgcn_s_sleep(2);
      }
    }
    __syncthreads();
    __threadfence();
  }
}

// ---------------------------------------------------------------------------
extern "C" void kernel_launch(void* const* d_in, const int* in_sizes, int n_in,
                              void* d_out, int out_size, void* d_ws, size_t ws_size,
                              hipStream_t stream) {
  (void)in_sizes; (void)n_in; (void)out_size; (void)ws_size;
  const float* x     = (const float*)d_in[0];
  const float* Wi    = (const float*)d_in[1];
  const float* Wh    = (const float*)d_in[2];
  const float* Bgf   = (const float*)d_in[3];
  const float* Wni   = (const float*)d_in[4];
  const float* Wnh   = (const float*)d_in[5];
  const float* Bnif  = (const float*)d_in[6];
  const float* Bnhf  = (const float*)d_in[7];
  const float* Winvi = (const float*)d_in[8];
  const float* Winvh = (const float*)d_in[9];
  const float* Bgb   = (const float*)d_in[10];
  const float* Wninvi= (const float*)d_in[11];
  const float* Wninvh= (const float*)d_in[12];
  const float* Bnib  = (const float*)d_in[13];
  const float* Bnhb  = (const float*)d_in[14];

  char* ws = (char*)d_ws;
  __bf16* xbf    = (__bf16*)(ws + OFF_XBF);
  __bf16* winp_f = (__bf16*)(ws + OFF_WIF);
  __bf16* wrec_f = (__bf16*)(ws + OFF_WRF);
  __bf16* winp_b = (__bf16*)(ws + OFF_WIB);
  __bf16* wrec_b = (__bf16*)(ws + OFF_WRB);
  __bf16* hbf    = (__bf16*)(ws + OFF_HBF);
  unsigned* bar  = (unsigned*)(ws + OFF_BAR);

  float* out = (float*)d_out;                       // [32][1024][2048]
  float* ht  = out + (size_t)GB * GS * 2 * GH;      // [32][2048]

  // ---- prep: x -> bf16 ----
  gru_cvt_x<<<2048, 256, 0, stream>>>(x, xbf, (size_t)GB * GS * GK);

  // ---- prep: weights -> transposed bf16 (column groups r | z | n) ----
  gru_pack_w<<<2048, 256, 0, stream>>>(Wi,  winp_f,                   GK, 2*GH, 2*GH);
  gru_pack_w<<<1024, 256, 0, stream>>>(Wni, winp_f + (size_t)2*GH*GK, GK, GH,   GH);
  gru_pack_w<<<2048, 256, 0, stream>>>(Wh,  wrec_f,                   GH, 2*GH, 2*GH);
  gru_pack_w<<<1024, 256, 0, stream>>>(Wnh, wrec_f + (size_t)2*GH*GH, GH, GH,   GH);
  gru_pack_w<<<2048, 256, 0, stream>>>(Winvi,  winp_b,                   GK, 2*GH, 2*GH);
  gru_pack_w<<<1024, 256, 0, stream>>>(Wninvi, winp_b + (size_t)2*GH*GK, GK, GH,   GH);
  gru_pack_w<<<2048, 256, 0, stream>>>(Winvh,  wrec_b,                   GH, 2*GH, 2*GH);
  gru_pack_w<<<1024, 256, 0, stream>>>(Wninvh, wrec_b + (size_t)2*GH*GH, GH, GH,   GH);

  // ---- prep: zero h bf16 double-buffers + reset barriers ----
  gru_init<<<256, 256, 0, stream>>>(hbf, bar);

  // ---- persistent bidirectional scan: 128 blocks x 2 waves, LDS weights ----
  gru_scan<<<128, 64, (size_t)LDS_ELEMS * 2, stream>>>(
      xbf, winp_f, wrec_f, winp_b, wrec_b,
      Bgf, Bnif, Bnhf, Bgb, Bnib, Bnhb,
      hbf, bar, out, ht);
}
